// Sequence_9113920602326
// MI455X (gfx1250) — matrix operand order repaired and checked by reading, not compiled, not run on hardware
//
#include <hip/hip_runtime.h>

#define T_STEPS 40
#define GATED   30
#define BATCH   4096
#define XD      150
#define HD      150
#define KREAL   300   // XD + HD
#define KP      320   // padded K: 10 chunks of 32
#define NREAL   600   // 4 * HD
#define NP      640   // padded N: 40 tiles of 16

typedef __attribute__((ext_vector_type(16))) _Float16 v16h;
typedef __attribute__((ext_vector_type(8)))  _Float16 v8h;
typedef __attribute__((ext_vector_type(8)))  float    v8f;

// ---- workspace layout (bytes) ----
#define WS_WF16 0                                   // f16 weights [NP][KP]
#define WS_BIAS (WS_WF16 + NP*KP*2)                 // f32 [NP]
#define WS_H    (WS_BIAS + 4096)                    // f32 [BATCH*HD]
#define WS_C    (WS_H + BATCH*HD*4)                 // f32 [BATCH*HD]
#define WS_G    (WS_C + BATCH*HD*4)                 // f32 [BATCH*NP]
#define WS_A    (WS_G + BATCH*NP*4)                 // f16 [BATCH*KP]

// ---- CDNA5 async Global->LDS copy (16B per lane, ASYNCcnt-tracked) ----
static __device__ __forceinline__ void async_copy_b128(uint32_t lds_byte_addr, const void* gaddr) {
  asm volatile("global_load_async_to_lds_b128 %0, %1, off"
               :: "v"(lds_byte_addr), "v"((unsigned long long)(uintptr_t)gaddr)
               : "memory");
}
static __device__ __forceinline__ void wait_async0() {
#if __has_builtin(__builtin_amdgcn_s_wait_asynccnt)
  __builtin_amdgcn_s_wait_asynccnt(0);
#else
  asm volatile("s_wait_asynccnt 0x0" ::: "memory");
#endif
}

// ---------------- weight repack: W[n][k] = [w_ih | w_hh], f16, zero-padded ----
__global__ void prep_weights(const float* __restrict__ w_ih, const float* __restrict__ w_hh,
                             const float* __restrict__ b_ih, const float* __restrict__ b_hh,
                             _Float16* __restrict__ wf, float* __restrict__ bias) {
  int i = blockIdx.x * 256 + threadIdx.x;
  if (i < NP * KP) {
    int n = i / KP, k = i - n * KP;
    float v = 0.0f;
    if (n < NREAL) {
      if (k < XD)         v = w_ih[n * XD + k];
      else if (k < KREAL) v = w_hh[n * HD + (k - XD)];
    }
    wf[i] = (_Float16)v;
  }
  if (i < NP) bias[i] = (i < NREAL) ? (b_ih[i] + b_hh[i]) : 0.0f;
}

__global__ void init_state(float* __restrict__ h, float* __restrict__ c) {
  int i = blockIdx.x * 256 + threadIdx.x;
  if (i < BATCH * HD) { h[i] = 0.0f; c[i] = 0.0f; }
}

// ---------------- per-step A panel pack: Af16[b][k] = f16([x_t | h | 0]) ----
__global__ __launch_bounds__(256)
void pack_A(const float* __restrict__ xt, const float* __restrict__ h,
            _Float16* __restrict__ A) {
  int i = blockIdx.x * 256 + threadIdx.x;          // over BATCH * KP/2 half2 pairs
  if (i >= BATCH * (KP / 2)) return;
  int b = i / (KP / 2), kp = i - b * (KP / 2);
  int k = kp * 2;
  float2 v = make_float2(0.0f, 0.0f);
  if (k < XD)          v = *(const float2*)(xt + b * XD + k);        // XD even: no straddle
  else if (k < KREAL)  v = *(const float2*)(h + b * HD + (k - XD));
  union { _Float16 q[2]; unsigned int u; } p;
  p.q[0] = (_Float16)v.x; p.q[1] = (_Float16)v.y;
  *(unsigned int*)(A + (size_t)b * KP + k) = p.u;
}

// ---------------- per-step GEMM: G = A @ W^T + bias  (f16 WMMA) ----
#define ROWS 64
#define LDK  328   // padded LDS row stride (halves): conflict-free b128 reads

__global__ __launch_bounds__(256)
void lstm_gemm(const _Float16* __restrict__ Af,
               const _Float16* __restrict__ wf, const float* __restrict__ bias,
               float* __restrict__ G) {
  __shared__ __align__(16) _Float16 sIn[ROWS * LDK];
  const int tid = threadIdx.x;
  const int rowbase = blockIdx.x * ROWS;

  // async DMA the 64x320 f16 A tile into LDS (with LDK padding), 16B per lane
  {
    const uint32_t sbase = (uint32_t)(uintptr_t)(&sIn[0]);
    const _Float16* Arow = Af + (size_t)rowbase * KP;
    for (int i = tid; i < ROWS * (KP / 8); i += 256) {
      int r = i / (KP / 8), c = i - r * (KP / 8);         // c: 16B chunk within row
      const void* g = Arow + r * KP + c * 8;
      uint32_t l = sbase + (uint32_t)((r * LDK + c * 8) * 2);
      async_copy_b128(l, g);
    }
    wait_async0();
  }
  __syncthreads();

  const int lane = tid & 31;
  const int wid  = tid >> 5;
  const int mgrp = wid & 1;        // 2 M-tiles per wave (32 rows)
  const int ngrp = wid >> 1;       // 4 n-groups, 5 N-tiles each
  const int col  = lane & 15;
  const int hi   = lane >> 4;      // half-wave selector

  const int mrow0 = mgrp * 32;
  const int n0    = blockIdx.y * 320 + ngrp * 80;   // blockIdx.y in {0,1} -> covers NP=640

  v8f acc[2][5];
  const v8f vzero = {0.f,0.f,0.f,0.f,0.f,0.f,0.f,0.f};
  #pragma unroll
  for (int mi = 0; mi < 2; ++mi)
    #pragma unroll
    for (int ni = 0; ni < 5; ++ni) acc[mi][ni] = vzero;

  #pragma unroll
  for (int kc = 0; kc < KP / 32; ++kc) {
    const int kb = kc * 32;
    // A fragments from LDS (16-bit A layout: lanes 0-15 hold K 0-7,16-23; lanes 16-31 hold K 8-15,24-31)
    v16h a[2];
    #pragma unroll
    for (int mi = 0; mi < 2; ++mi) {
      const _Float16* p = &sIn[(mrow0 + mi * 16 + col) * LDK + kb + hi * 8];
      union { v16h v; v8h q[2]; } ua;
      ua.q[0] = *(const v8h*)p;
      ua.q[1] = *(const v8h*)(p + 16);
      a[mi] = ua.v;
    }
    // B fragments from global (lanes 0-15: K 0-7,16-23 of column n; lanes 16-31: K 8-15,24-31)
    v16h b[5];
    #pragma unroll
    for (int ni = 0; ni < 5; ++ni) {
      const _Float16* p = wf + (n0 + ni * 16 + col) * KP + kb + hi * 8;
      union { v16h v; v8h q[2]; } ub;
      ub.q[0] = *(const v8h*)p;
      ub.q[1] = *(const v8h*)(p + 16);
      b[ni] = ub.v;
    }
    #pragma unroll
    for (int mi = 0; mi < 2; ++mi)
      #pragma unroll
      for (int ni = 0; ni < 5; ++ni)
        acc[mi][ni] = __builtin_amdgcn_wmma_f32_16x16x32_f16(
            false, a[mi], false, b[ni], (short)0, acc[mi][ni], false, false);
  }

  // bias + store G (C/D layout: vgpr r -> row r + 8*hi, lane&15 -> col)
  #pragma unroll
  for (int ni = 0; ni < 5; ++ni) {
    const float bv = bias[n0 + ni * 16 + col];
    #pragma unroll
    for (int mi = 0; mi < 2; ++mi) {
      const int gr = rowbase + mrow0 + mi * 16 + hi * 8;
      float* gp = G + (size_t)gr * NP + n0 + ni * 16 + col;
      #pragma unroll
      for (int r = 0; r < 8; ++r)
        gp[(size_t)r * NP] = acc[mi][ni][r] + bv;
    }
  }
}

// ---------------- elementwise LSTM cell update ----
__global__ __launch_bounds__(256)
void lstm_cell(const float* __restrict__ G, float* __restrict__ h, float* __restrict__ c,
               float* __restrict__ outputs_t, float* __restrict__ outh, int writeOuth) {
  int idx = blockIdx.x * 256 + threadIdx.x;
  if (idx >= BATCH * HD) return;
  int b = idx / HD, j = idx - b * HD;
  const float* g = G + (size_t)b * NP;
  float gi = g[j];
  float gf = g[HD + j];
  float gg = g[2 * HD + j];
  float go = g[3 * HD + j];
  float cc = c[idx];
  float si = 1.0f / (1.0f + __expf(-gi));
  float sf = 1.0f / (1.0f + __expf(-gf));
  float so = 1.0f / (1.0f + __expf(-go));
  float cn = sf * cc + si * tanhf(gg);
  float hn = so * tanhf(cn);
  c[idx] = cn;
  h[idx] = hn;
  outputs_t[idx] = hn;
  if (writeOuth) outh[idx] = hn;
}

// ---------------- stop-gate MLP: softmax(relu(W3 relu(W2 relu(W1 [h;x]+b1)+b2)+b3)) ----
__global__ __launch_bounds__(256)
void stop_gate(const float* __restrict__ h, const float* __restrict__ xt,
               const float* __restrict__ w1, const float* __restrict__ b1,
               const float* __restrict__ w2, const float* __restrict__ b2,
               const float* __restrict__ w3, const float* __restrict__ b3,
               float* __restrict__ y) {
  __shared__ float sin_[4][304];
  __shared__ float s1[4][52];
  __shared__ float s2[4][28];
  __shared__ float s3[4][2];
  const int tid = threadIdx.x;
  const int lr  = tid >> 6;        // 4 rows per block
  const int tj  = tid & 63;
  const int row = blockIdx.x * 4 + lr;

  for (int k = tj; k < KREAL; k += 64)
    sin_[lr][k] = (k < HD) ? h[row * HD + k] : xt[row * XD + (k - HD)];
  __syncthreads();

  if (tj < 50) {
    float a = b1[tj];
    const float* wr = w1 + tj * KREAL;
    for (int k = 0; k < KREAL; ++k) a += wr[k] * sin_[lr][k];
    s1[lr][tj] = fmaxf(a, 0.0f);
  }
  __syncthreads();
  if (tj < 25) {
    float a = b2[tj];
    const float* wr = w2 + tj * 50;
    for (int k = 0; k < 50; ++k) a += wr[k] * s1[lr][k];
    s2[lr][tj] = fmaxf(a, 0.0f);
  }
  __syncthreads();
  if (tj < 2) {
    float a = b3[tj];
    const float* wr = w3 + tj * 25;
    for (int k = 0; k < 25; ++k) a += wr[k] * s2[lr][k];
    s3[lr][tj] = fmaxf(a, 0.0f);
  }
  __syncthreads();
  if (tj == 0) {
    float a = s3[lr][0], bb = s3[lr][1];
    float m = fmaxf(a, bb);
    float e0 = __expf(a - m), e1 = __expf(bb - m);
    float s = e0 + e1;
    y[row * 2 + 0] = e0 / s;
    y[row * 2 + 1] = e1 / s;
  }
}

extern "C" void kernel_launch(void* const* d_in, const int* in_sizes, int n_in,
                              void* d_out, int out_size, void* d_ws, size_t ws_size,
                              hipStream_t stream) {
  (void)in_sizes; (void)n_in; (void)out_size; (void)ws_size;
  const float* x    = (const float*)d_in[0];
  const float* w_ih = (const float*)d_in[1];
  const float* w_hh = (const float*)d_in[2];
  const float* b_ih = (const float*)d_in[3];
  const float* b_hh = (const float*)d_in[4];
  const float* w1   = (const float*)d_in[5];
  const float* b1   = (const float*)d_in[6];
  const float* w2   = (const float*)d_in[7];
  const float* b2   = (const float*)d_in[8];
  const float* w3   = (const float*)d_in[9];
  const float* b3   = (const float*)d_in[10];

  float* out = (float*)d_out;
  char*  ws  = (char*)d_ws;
  _Float16* wf   = (_Float16*)(ws + WS_WF16);
  float*    bias = (float*)(ws + WS_BIAS);
  float*    hs   = (float*)(ws + WS_H);
  float*    cs   = (float*)(ws + WS_C);
  float*    G    = (float*)(ws + WS_G);
  _Float16* Af   = (_Float16*)(ws + WS_A);

  float* outh    = out;
  float* outputs = out + BATCH * HD;
  float* ys      = out + BATCH * HD + (size_t)T_STEPS * BATCH * HD;

  prep_weights<<<(NP * KP + 255) / 256, 256, 0, stream>>>(w_ih, w_hh, b_ih, b_hh, wf, bias);
  init_state<<<(BATCH * HD + 255) / 256, 256, 0, stream>>>(hs, cs);

  for (int t = 0; t < T_STEPS; ++t) {
    const float* xt = x + (size_t)t * BATCH * XD;
    pack_A<<<(BATCH * (KP / 2) + 255) / 256, 256, 0, stream>>>(xt, hs, Af);
    lstm_gemm<<<dim3(BATCH / ROWS, 2), 256, 0, stream>>>(Af, wf, bias, G);
    lstm_cell<<<(BATCH * HD + 255) / 256, 256, 0, stream>>>(
        G, hs, cs, outputs + (size_t)t * BATCH * HD, outh, (t == GATED - 1) ? 1 : 0);
    if (t < GATED)
      stop_gate<<<BATCH / 4, 256, 0, stream>>>(hs, xt, w1, b1, w2, b2, w3, b3,
                                               ys + (size_t)t * BATCH * 2);
  }
}
